// InterestEvolvingLayer_42820823941608
// MI455X (gfx1250) — matrix operand order repaired and checked by reading, not compiled, not run on hardware
//
#include <hip/hip_runtime.h>
#include <hip/hip_bf16.h>

typedef __attribute__((ext_vector_type(16))) _Float16 v16h;
typedef __attribute__((ext_vector_type(8)))  float    v8f;

#define B_SZ 1024
#define T_SZ 200

#if defined(__AMDGCN__) && __has_builtin(__builtin_amdgcn_global_load_async_to_lds_b32)
#define HAVE_ASYNC_LDS 1
#endif

__device__ static inline v8f wmma_f16(v16h a, v16h b, v8f c) {
  return __builtin_amdgcn_wmma_f32_16x16x32_f16(
      /*neg_a=*/false, a, /*neg_b=*/false, b,
      /*c_mod=*/(short)0, c, /*reuse_a=*/false, /*reuse_b=*/false);
}

// WMMA A-fragment swizzle (this code's contiguous k order, 16-bit A 16x32):
// lane l = (hf<<4)|m reads element i at k = hf*16 + i.
// Inverse: hf=ek>>4, i=ek&15; flat = ((hf<<4)|m)*16 + i.
// Note: for even ek, (ek, ek+1) map to contiguous halves (i, i+1).
__device__ static inline int a_sw_idx(int m, int ek) {
  int hf = ek >> 4;
  int i  = ek & 15;
  return (((hf << 4) | m) << 4) + i;
}

__device__ static inline float sigmoidf_(float x) {
  return 1.0f / (1.0f + __expf(-x));
}

// ---------------- prep: repack f32 weights into f16 WMMA fragments ----------

__global__ void prep_attw(const float* __restrict__ w1, const float* __restrict__ w2,
                          _Float16* __restrict__ WKf, _Float16* __restrict__ WDf,
                          _Float16* __restrict__ W2Pf) {
  int o = blockIdx.x * 256 + threadIdx.x;
  if (o < 4 * 5 * 512) {                    // K=128 (4 kt), N=80 (5 nt)
    int f = o >> 9, l = (o >> 4) & 31, i = o & 15;
    int kt = f / 5, nt = f % 5;
    int k = kt * 32 + ((l >> 4) << 4) + i;
    int n = nt * 16 + (l & 15);
    WKf[o] = (_Float16)(w1[(128 + k) * 80 + n] - w1[(256 + k) * 80 + n]);
    WDf[o] = (_Float16)(w1[(384 + k) * 80 + n]);
  }
  if (o < 3 * 3 * 512) {                    // K=96 (3 kt), N=48 (3 nt), padded
    int f = o >> 9, l = (o >> 4) & 31, i = o & 15;
    int kt = f / 3, nt = f % 3;
    int k = kt * 32 + ((l >> 4) << 4) + i;
    int n = nt * 16 + (l & 15);
    W2Pf[o] = (_Float16)((k < 80 && n < 40) ? w2[k * 40 + n] : 0.0f);
  }
}

// r-gate (cols 0..127) and h-gate (cols 256..383) tiles only; 16 n-tiles.
__global__ void prep_rnnw(const float* __restrict__ w_ih, const float* __restrict__ w_hh,
                          _Float16* __restrict__ WIHTf, _Float16* __restrict__ WHHTf) {
  int o = blockIdx.x * 256 + threadIdx.x;
  if (o >= 4 * 16 * 512) return;
  int f = o >> 9, l = (o >> 4) & 31, i = o & 15;
  int kt = f / 16, nt = f % 16;
  int e  = kt * 32 + ((l >> 4) << 4) + i;
  int gn = ((nt < 8) ? nt * 16 : 256 + (nt - 8) * 16) + (l & 15);
  WIHTf[o] = (_Float16)w_ih[gn * 128 + e];
  WHHTf[o] = (_Float16)w_hh[gn * 128 + e];
}

// keys f32 -> f16 (plain layout), streamed once
__global__ void prep_keysh(const float* __restrict__ keys, _Float16* __restrict__ keysh) {
  size_t i = (size_t)blockIdx.x * 2048 + threadIdx.x;
#pragma unroll
  for (int s = 0; s < 8; ++s, i += 256) keysh[i] = (_Float16)keys[i];
}

// QW[b][n] = sum_e q[b][e] * (w1[e][n] + w1[256+e][n])
__global__ void qw_kernel(const float* __restrict__ q, const float* __restrict__ w1,
                          float* __restrict__ QW) {
  int i = blockIdx.x * 256 + threadIdx.x;
  if (i >= B_SZ * 80) return;
  int b = i / 80, n = i % 80;
  float s = 0.0f;
  for (int e = 0; e < 128; ++e)
    s += q[b * 128 + e] * (w1[e * 80 + n] + w1[(256 + e) * 80 + n]);
  QW[i] = s;
}

// ---------------- attention MLP -------------------------------------------

__global__ void __launch_bounds__(128)
att_mlp_kernel(const float* __restrict__ keys, const float* __restrict__ queries,
               const float* __restrict__ QW,
               const _Float16* __restrict__ WKf, const _Float16* __restrict__ WDf,
               const _Float16* __restrict__ W2Pf,
               const float* __restrict__ b1, const float* __restrict__ b2,
               const float* __restrict__ wd, const float* __restrict__ bd,
               float* __restrict__ scores) {
  __shared__ _Float16 sA1[4][4][512];   // [wave][kt][lane*16+i]  keys f16
  __shared__ _Float16 sA2[4][4][512];   // q*k f16
  __shared__ _Float16 sH1[4][3][512];   // layer-1 act, swizzled, K padded 96
  __shared__ _Float16 sH2[4][16][48];   // layer-2 act, plain

  const int tid  = threadIdx.x;
  const int w    = tid >> 5;
  const int lane = tid & 31;
  const int hf   = lane >> 4;
  const int r0   = blockIdx.x * 64 + w * 16;   // flattened (b*T+t) row base

  for (int idx = lane; idx < 3 * 512; idx += 32) (&sH1[w][0][0])[idx] = (_Float16)0.0f;

  for (int m = 0; m < 16; ++m) {
    const int r = r0 + m;
    const float* kp = keys + (size_t)r * 128;
    const float* qp = queries + (size_t)(r / T_SZ) * 128;
    for (int e = lane; e < 128; e += 32) {
      float kv = kp[e], qv = qp[e];
      int pos = a_sw_idx(m, e & 31);
      sA1[w][e >> 5][pos] = (_Float16)kv;
      sA2[w][e >> 5][pos] = (_Float16)(qv * kv);
    }
  }
  __syncthreads();

  // layer 1: h1 = sigmoid(k@WK + (q*k)@WD + QW[b] + b1), N=80
  for (int nt = 0; nt < 5; ++nt) {
    v8f acc = {};
#pragma unroll
    for (int kt = 0; kt < 4; ++kt) {
      v16h a1 = *(const v16h*)&sA1[w][kt][lane * 16];
      v16h bk = *(const v16h*)(WKf + (size_t)(kt * 5 + nt) * 512 + lane * 16);
      acc = wmma_f16(a1, bk, acc);
      v16h a2 = *(const v16h*)&sA2[w][kt][lane * 16];
      v16h bq = *(const v16h*)(WDf + (size_t)(kt * 5 + nt) * 512 + lane * 16);
      acc = wmma_f16(a2, bq, acc);
    }
    const int n   = nt * 16 + (lane & 15);
    const int kt2 = n >> 5, ek2 = n & 31;
#pragma unroll
    for (int v = 0; v < 8; ++v) {
      const int m = v + hf * 8;
      const int b = (r0 + m) / T_SZ;
      float x = acc[v] + QW[b * 80 + n] + b1[n];
      sH1[w][kt2][a_sw_idx(m, ek2)] = (_Float16)sigmoidf_(x);
    }
  }
  __syncthreads();

  // layer 2: h2 = sigmoid(h1@W2 + b2), N=40 (padded 48), K=96
#pragma unroll
  for (int nt = 0; nt < 3; ++nt) {
    v8f acc = {};
#pragma unroll
    for (int kt = 0; kt < 3; ++kt) {
      v16h a  = *(const v16h*)&sH1[w][kt][lane * 16];
      v16h bf = *(const v16h*)(W2Pf + (size_t)(kt * 3 + nt) * 512 + lane * 16);
      acc = wmma_f16(a, bf, acc);
    }
    const int n = nt * 16 + (lane & 15);
    if (n < 40) {
#pragma unroll
      for (int v = 0; v < 8; ++v)
        sH2[w][v + hf * 8][n] = (_Float16)sigmoidf_(acc[v] + b2[n]);
    }
  }
  __syncthreads();

  // layer 3: score = h2 . wd + bd
  if (lane < 16) {
    const int m = lane;
    float s = bd[0];
#pragma unroll
    for (int j = 0; j < 40; ++j) s += (float)sH2[w][m][j] * wd[j];
    scores[r0 + m] = s;
  }
}

// ---------------- masked softmax over T (in place) --------------------------

__global__ void __launch_bounds__(256)
softmax_kernel(const int* __restrict__ keys_length, float* __restrict__ scores) {
  __shared__ float red[256];
  const int b = blockIdx.x, tid = threadIdx.x;
  const int L = keys_length[b];
  const float scale = 0.08838834764831845f;  // 1/sqrt(128)
  float myval = -__builtin_inff();
  if (tid < T_SZ && tid < L) myval = scores[b * T_SZ + tid] * scale;
  red[tid] = myval;
  __syncthreads();
  for (int s = 128; s > 0; s >>= 1) {
    if (tid < s) red[tid] = fmaxf(red[tid], red[tid + s]);
    __syncthreads();
  }
  const float mx = red[0];
  __syncthreads();
  float e = (tid < T_SZ && tid < L) ? __expf(myval - mx) : 0.0f;
  red[tid] = e;
  __syncthreads();
  for (int s = 128; s > 0; s >>= 1) {
    if (tid < s) red[tid] += red[tid + s];
    __syncthreads();
  }
  const float sum = red[0];
  __syncthreads();
  if (tid < T_SZ) scores[b * T_SZ + tid] = e / sum;
}

// ---------------- AGRU recurrence -------------------------------------------
// Weights in VGPRs; keys_t async-DMA'd into swizzled LDS fragments (gfx1250
// GLOBAL_LOAD_ASYNC_TO_LDS_B32, ASYNCcnt) when the builtin is available.

__global__ void __launch_bounds__(256)
agru_kernel(const _Float16* __restrict__ keysh, const float* __restrict__ att,
            const int* __restrict__ keys_length,
            const _Float16* __restrict__ WIHTf, const _Float16* __restrict__ WHHTf,
            const float* __restrict__ b_ih, const float* __restrict__ b_hh,
            float* __restrict__ out) {
  __shared__ float    sH[16][128];       // f32 hidden state
  __shared__ _Float16 sHsw[4][512];      // h as swizzled f16 A-fragments
  __shared__ _Float16 sKsw[4][512];      // keys_t swizzled f16
  __shared__ float    sGI[16][256];      // gi: [r-gate | h-gate]
  __shared__ float    sGH[16][256];      // gh: [r-gate | h-gate]

  const int tid  = threadIdx.x;
  const int w    = tid >> 5;
  const int lane = tid & 31;
  const int hf   = lane >> 4;
  const int b0   = blockIdx.x * 16;

  // hoist this wave's weight fragments into VGPRs (invariant over t)
  v16h bI[2][4], bH[2][4];
#pragma unroll
  for (int part = 0; part < 2; ++part) {
    const int ntile = part ? 8 + w : w;
#pragma unroll
    for (int kt = 0; kt < 4; ++kt) {
      const size_t off = (size_t)(kt * 16 + ntile) * 512 + lane * 16;
      bI[part][kt] = *(const v16h*)(WIHTf + off);
      bH[part][kt] = *(const v16h*)(WHHTf + off);
    }
  }

  // per-thread row mapping for the gate-update stage; hoist invariants
  const int um  = tid >> 4;   // row 0..15
  const int uj0 = tid & 15;   // j = uj0 + s*16
  float bihr[8], bihh[8], bhhr[8], bhhh[8];
#pragma unroll
  for (int s = 0; s < 8; ++s) {
    const int j = uj0 + s * 16;
    bihr[s] = b_ih[j]; bihh[s] = b_ih[256 + j];
    bhhr[s] = b_hh[j]; bhhh[s] = b_hh[256 + j];
  }
  const int myL = keys_length[b0 + um];

  for (int idx = tid; idx < 2048; idx += 256) sH[idx >> 7][idx & 127] = 0.0f;
  __syncthreads();

  for (int t = 0; t < T_SZ; ++t) {
#ifdef HAVE_ASYNC_LDS
    // keys_t tile: 1024 async b32 transfers straight into swizzled LDS
#pragma unroll
    for (int s = 0; s < 4; ++s) {
      const int p = tid + s * 256;       // pair index 0..1023
      const int m = p >> 6;              // row
      const int e = (p & 63) * 2;        // even element -> contiguous halves
      const int pos = a_sw_idx(m, e & 31);
      __builtin_amdgcn_global_load_async_to_lds_b32(
          (__attribute__((address_space(1))) int*)
              (keysh + ((size_t)(b0 + m) * T_SZ + t) * 128 + e),
          (__attribute__((address_space(3))) int*)(&sKsw[e >> 5][pos]),
          0, 0);
    }
    for (int idx = tid; idx < 2048; idx += 256) {
      const int m = idx >> 7, e = idx & 127;
      sHsw[e >> 5][a_sw_idx(m, e & 31)] = (_Float16)sH[m][e];
    }
#if __has_builtin(__builtin_amdgcn_s_wait_asynccnt)
    __builtin_amdgcn_s_wait_asynccnt(0);
#else
    asm volatile("s_wait_asynccnt 0x0" ::: "memory");
#endif
#else
    for (int idx = tid; idx < 2048; idx += 256) {
      const int m = idx >> 7, e = idx & 127;
      const int pos = a_sw_idx(m, e & 31);
      sKsw[e >> 5][pos] = keysh[((size_t)(b0 + m) * T_SZ + t) * 128 + e];
      sHsw[e >> 5][pos] = (_Float16)sH[m][e];
    }
#endif
    if (t + 1 < T_SZ)
      __builtin_prefetch(&keysh[((size_t)(b0 + (tid >> 4)) * T_SZ + t + 1) * 128], 0, 1);
    __syncthreads();

#pragma unroll
    for (int part = 0; part < 2; ++part) {
      v8f ai = {}, ah = {};
#pragma unroll
      for (int kt = 0; kt < 4; ++kt) {
        v16h ak = *(const v16h*)&sKsw[kt][lane * 16];
        ai = wmma_f16(ak, bI[part][kt], ai);
        v16h am = *(const v16h*)&sHsw[kt][lane * 16];
        ah = wmma_f16(am, bH[part][kt], ah);
      }
      const int nl = part * 128 + w * 16 + (lane & 15);
#pragma unroll
      for (int v = 0; v < 8; ++v) {
        const int m = v + hf * 8;
        sGI[m][nl] = ai[v];
        sGH[m][nl] = ah[v];
      }
    }
    __syncthreads();

    // gate math + state update (each thread owns row um, 8 j's)
    {
      const float a = att[(size_t)(b0 + um) * T_SZ + t];
      const bool valid = t < myL;
#pragma unroll
      for (int s = 0; s < 8; ++s) {
        const int j = uj0 + s * 16;
        const float ir  = sGI[um][j]       + bihr[s];
        const float ihv = sGI[um][128 + j] + bihh[s];
        const float hr  = sGH[um][j]       + bhhr[s];
        const float hh  = sGH[um][128 + j] + bhhh[s];
        const float r   = sigmoidf_(ir + hr);
        const float nh  = tanhf(ihv + r * hh);
        const float h   = sH[um][j];
        const float hy  = (1.0f - a) * h + a * nh;
        if (valid) sH[um][j] = hy;
      }
    }
    __syncthreads();
  }

  for (int idx = tid; idx < 2048; idx += 256) {
    const int m = idx >> 7, j = idx & 127;
    out[(size_t)(b0 + m) * 128 + j] = sH[m][j];
  }
}

// ---------------- launch ----------------------------------------------------

extern "C" void kernel_launch(void* const* d_in, const int* in_sizes, int n_in,
                              void* d_out, int out_size, void* d_ws, size_t ws_size,
                              hipStream_t stream) {
  const float* queries = (const float*)d_in[0];
  const float* keys    = (const float*)d_in[1];
  const int*   klen    = (const int*)d_in[2];
  const float* w1      = (const float*)d_in[3];
  const float* b1      = (const float*)d_in[4];
  const float* w2      = (const float*)d_in[5];
  const float* b2      = (const float*)d_in[6];
  const float* wd      = (const float*)d_in[7];
  const float* bdv     = (const float*)d_in[8];
  const float* w_ih    = (const float*)d_in[9];
  const float* w_hh    = (const float*)d_in[10];
  const float* b_ih    = (const float*)d_in[11];
  const float* b_hh    = (const float*)d_in[12];
  float* out = (float*)d_out;

  char* ws = (char*)d_ws;
  float*    scores = (float*)(ws + 0);            // B*T f32      819200 B
  float*    QW     = (float*)(ws + 819200);       // B*80 f32     327680 B
  _Float16* WKf    = (_Float16*)(ws + 1146880);   // 20 frags     20480 B
  _Float16* WDf    = (_Float16*)(ws + 1167360);   // 20 frags     20480 B
  _Float16* W2Pf   = (_Float16*)(ws + 1187840);   // 9 frags       9216 B
  _Float16* WIHTf  = (_Float16*)(ws + 1197056);   // 64 frags     65536 B
  _Float16* WHHTf  = (_Float16*)(ws + 1262592);   // 64 frags     65536 B
  _Float16* keysh  = (_Float16*)(ws + 1328128);   // B*T*E f16  52428800 B

  hipLaunchKernelGGL(prep_attw,  dim3(40),    dim3(256), 0, stream, w1, w2, WKf, WDf, W2Pf);
  hipLaunchKernelGGL(prep_rnnw,  dim3(128),   dim3(256), 0, stream, w_ih, w_hh, WIHTf, WHHTf);
  hipLaunchKernelGGL(prep_keysh, dim3(12800), dim3(256), 0, stream, keys, keysh);
  hipLaunchKernelGGL(qw_kernel,  dim3(320),   dim3(256), 0, stream, queries, w1, QW);

  hipLaunchKernelGGL(att_mlp_kernel, dim3((B_SZ * T_SZ) / 64), dim3(128), 0, stream,
                     keys, queries, QW, WKf, WDf, W2Pf, b1, b2, wd, bdv, scores);
  hipLaunchKernelGGL(softmax_kernel, dim3(B_SZ), dim3(256), 0, stream, klen, scores);
  hipLaunchKernelGGL(agru_kernel, dim3(B_SZ / 16), dim3(256), 0, stream,
                     keysh, scores, klen, WIHTf, WHHTf, b_ih, b_hh, out);
}
